// GumbelVectorQuantizer_36249523978734
// MI455X (gfx1250) — matrix operand order repaired and checked by reading, not compiled, not run on hardware
//
#include <hip/hip_runtime.h>
#include <hip/hip_bf16.h>

typedef __attribute__((ext_vector_type(16))) __bf16 v16bf;
typedef __attribute__((ext_vector_type(8)))  __bf16 v8bf;
typedef __attribute__((ext_vector_type(8)))  float  v8f;

#define M_TOK   16384   // B*T
#define K_F     512
#define N_GE    640     // G*E
#define E_      320
#define D_      384
#define GD_     768     // G*D

// ---------------------------------------------------------------------------
// Kernel 1: split fp32 -> bf16 hi + bf16 lo (near-fp32 precision for WMMA)
// ---------------------------------------------------------------------------
__global__ void split_bf16_kernel(const float* __restrict__ src,
                                  __bf16* __restrict__ hi,
                                  __bf16* __restrict__ lo, int count) {
    int i = blockIdx.x * blockDim.x + threadIdx.x;
    if (i < count) {
        float v = src[i];
        __bf16 h = (__bf16)v;
        float r = v - (float)h;
        hi[i] = h;
        lo[i] = (__bf16)r;
    }
}

// ---------------------------------------------------------------------------
// Kernel 2: logits[M,640] = x[M,512] @ W[640,512]^T + b   via bf16 WMMA.
// One wave computes a 16x64 tile (4 f32 accumulators): A fragments loaded
// once per k-step and reused across 4 column tiles -> 20 loads / 12 WMMAs.
// All 8 waves of a block share the same 4 column tiles (B hits L1).
// hi/lo bf16 split: hi*hi + hi*lo + lo*hi  (~2^-17 relative error vs fp32).
// ---------------------------------------------------------------------------
__global__ __launch_bounds__(256)
void gemm_logits_wmma(const __bf16* __restrict__ xhi, const __bf16* __restrict__ xlo,
                      const __bf16* __restrict__ whi, const __bf16* __restrict__ wlo,
                      const float*  __restrict__ bias,
                      float* __restrict__ logits) {
    const int lane = threadIdx.x & 31;
    const int wv   = threadIdx.x >> 5;          // 0..7
    const int l15  = lane & 15;
    const int grp  = lane >> 4;

    const int COLG = (N_GE / 16) / 4;           // 10 groups of 4 column tiles
    const int colg = blockIdx.x % COLG;
    const int tm   = (blockIdx.x / COLG) * 8 + wv;   // row tile (exact fit)
    const int tn0  = colg * 4;                       // first of 4 column tiles

    // A fragment source: row m, two 8-element (16B) chunks per k-step
    const size_t arow = (size_t)(tm * 16 + l15) * K_F;
    // B fragment sources: W rows (W^T columns), two 8-element chunks per k-step
    size_t brow[4];
#pragma unroll
    for (int i = 0; i < 4; ++i)
        brow[i] = (size_t)((tn0 + i) * 16 + l15) * K_F;

    v8f acc0 = {}, acc1 = {}, acc2 = {}, acc3 = {};

    for (int kk = 0; kk < K_F; kk += 32) {
        const int ab = kk + grp * 8;            // A: VGPR0-3 ks, VGPR4-7 at +16
        const int bb = kk + grp * 8;            // B: VGPR0-3 ks, VGPR4-7 at +16

        // prefetch next k-tile of the streamed A operand
        if (kk + 32 < K_F) {
            __builtin_prefetch(xhi + arow + ab + 32, 0, 1);
            __builtin_prefetch(xlo + arow + ab + 32, 0, 1);
        }

        v8bf ah0 = *(const v8bf*)(xhi + arow + ab);
        v8bf ah1 = *(const v8bf*)(xhi + arow + ab + 16);
        v8bf al0 = *(const v8bf*)(xlo + arow + ab);
        v8bf al1 = *(const v8bf*)(xlo + arow + ab + 16);
        v16bf ah = __builtin_shufflevector(ah0, ah1, 0,1,2,3,4,5,6,7,8,9,10,11,12,13,14,15);
        v16bf al = __builtin_shufflevector(al0, al1, 0,1,2,3,4,5,6,7,8,9,10,11,12,13,14,15);

#pragma unroll
        for (int i = 0; i < 4; ++i) {
            v8bf bh0 = *(const v8bf*)(whi + brow[i] + bb);
            v8bf bh1 = *(const v8bf*)(whi + brow[i] + bb + 16);
            v8bf bl0 = *(const v8bf*)(wlo + brow[i] + bb);
            v8bf bl1 = *(const v8bf*)(wlo + brow[i] + bb + 16);
            v16bf bh = __builtin_shufflevector(bh0, bh1, 0,1,2,3,4,5,6,7,8,9,10,11,12,13,14,15);
            v16bf bl = __builtin_shufflevector(bl0, bl1, 0,1,2,3,4,5,6,7,8,9,10,11,12,13,14,15);

            v8f* acc = (i == 0) ? &acc0 : (i == 1) ? &acc1 : (i == 2) ? &acc2 : &acc3;
            *acc = __builtin_amdgcn_wmma_f32_16x16x32_bf16(false, ah, false, bh, (short)0, *acc, false, false);
            *acc = __builtin_amdgcn_wmma_f32_16x16x32_bf16(false, ah, false, bl, (short)0, *acc, false, false);
            *acc = __builtin_amdgcn_wmma_f32_16x16x32_bf16(false, al, false, bh, (short)0, *acc, false, false);
        }
    }

    // C/D layout: VGPR r -> row (tm*16 + grp*8 + r), col = tn*16 + (lane&15)
    const size_t outrow = (size_t)(tm * 16 + grp * 8) * N_GE;
#pragma unroll
    for (int i = 0; i < 4; ++i) {
        const int ncol = (tn0 + i) * 16 + l15;
        const float bv = bias[ncol];
        const v8f acc = (i == 0) ? acc0 : (i == 1) ? acc1 : (i == 2) ? acc2 : acc3;
        float* out = logits + outrow + ncol;
#pragma unroll
        for (int r = 0; r < 8; ++r)
            out[(size_t)r * N_GE] = acc[r] + bv;
    }
}

// ---------------------------------------------------------------------------
// Kernel 3: per-row (token,g) pass. 1 wave per row of 320 logits.
// Computes: hard argmax (code ppl), softmax partials (prob ppl),
// gumbel-softmax straight-through y -> cb, and quantized = gather(entries).
// Deterministic: LDS per-wave partials reduced in fixed order.
// ---------------------------------------------------------------------------
__global__ __launch_bounds__(1024)
void row_kernel(const float* __restrict__ logits, const float* __restrict__ gumbel,
                const float* __restrict__ entries,
                float* __restrict__ out_q, float* __restrict__ out_cb,
                float* __restrict__ partials) {
    __shared__ float s_part[32 * E_];   // per-wave softmax probs (40 KB)
    __shared__ int   s_idx[32];         // per-wave hard argmax

    const int t    = threadIdx.x;
    const int lane = t & 31;
    const int wv   = t >> 5;
    const int row  = blockIdx.x * 32 + wv;   // grid covers exactly 32768 rows
    const int token = row >> 1;
    const int g     = row & 1;

    const float* lrow = logits + (size_t)token * N_GE + (size_t)g * E_;
    const float* grow = gumbel + (size_t)row * E_;

    float lv[10], gv[10];
#pragma unroll
    for (int j = 0; j < 10; ++j) {
        int e = lane + j * 32;
        lv[j] = lrow[e];
        gv[j] = (lv[j] + grow[e]) * 0.5f;    // (logits+gumbel)/TAU, TAU=2
    }

    // lane-local max/argmax (strict > keeps lowest index, matching jnp.argmax)
    float m1 = lv[0]; int a1 = lane;
    float m2 = gv[0]; int a2 = lane;
#pragma unroll
    for (int j = 1; j < 10; ++j) {
        int e = lane + j * 32;
        if (lv[j] > m1) { m1 = lv[j]; a1 = e; }
        if (gv[j] > m2) { m2 = gv[j]; a2 = e; }
    }
    // cross-lane reduce, tie -> lower index
#pragma unroll
    for (int off = 16; off; off >>= 1) {
        float ov = __shfl_xor(m1, off, 32); int oi = __shfl_xor(a1, off, 32);
        if (ov > m1 || (ov == m1 && oi < a1)) { m1 = ov; a1 = oi; }
        float ow = __shfl_xor(m2, off, 32); int oj = __shfl_xor(a2, off, 32);
        if (ow > m2 || (ow == m2 && oj < a2)) { m2 = ow; a2 = oj; }
    }

    // softmax sums
    float p[10], q[10];
    float ps = 0.f, qs = 0.f;
#pragma unroll
    for (int j = 0; j < 10; ++j) {
        p[j] = expf(lv[j] - m1); ps += p[j];
        q[j] = expf(gv[j] - m2); qs += q[j];
    }
#pragma unroll
    for (int off = 16; off; off >>= 1) {
        ps += __shfl_xor(ps, off, 32);
        qs += __shfl_xor(qs, off, 32);
    }
    const float pinv = 1.f / ps, qinv = 1.f / qs;

    // straight-through y (exactly one-hot except argmax entry), cb output
    float* cbrow = out_cb + (size_t)token * N_GE + (size_t)g * E_;
    float yv_local = 0.f;
#pragma unroll
    for (int j = 0; j < 10; ++j) {
        int e = lane + j * 32;
        s_part[wv * E_ + e] = p[j] * pinv;
        float ys = q[j] * qinv;
        float y  = (e == a2) ? ((1.0f - ys) + ys) : ((0.0f - ys) + ys);
        cbrow[e] = y;
        if (e == a2) yv_local = y;
    }
    if (lane == 0) s_idx[wv] = a1;
    const float yv = __shfl(yv_local, a2 & 31, 32);

    // quantized = yv * entries[g, a2, :]  (exact: all other y entries are +0)
    const float* erow = entries + ((size_t)g * E_ + a2) * D_;
    float* qrow = out_q + (size_t)token * GD_ + (size_t)g * D_;
#pragma unroll
    for (int j = 0; j < 12; ++j)
        qrow[lane + j * 32] = yv * erow[lane + j * 32];

    __syncthreads();

    // fixed-order block reduction -> per-block partials (deterministic)
    if (t < 2 * E_) {
        int gg = t / E_, e = t - gg * E_;
        float asum = 0.f, hcnt = 0.f;
#pragma unroll
        for (int w = 0; w < 16; ++w) {
            int wvi = gg + 2 * w;                 // waves with matching g
            asum += s_part[wvi * E_ + e];
            hcnt += (s_idx[wvi] == e) ? 1.0f : 0.0f;
        }
        float* blk = partials + (size_t)blockIdx.x * 1280;
        blk[t]       = asum;   // [0,640): soft-prob partial sums
        blk[640 + t] = hcnt;   // [640,1280): hard one-hot counts
    }
}

// ---------------------------------------------------------------------------
// Kernel 4: fixed-order reduction of per-block partials -> two perplexities
// ---------------------------------------------------------------------------
__global__ void finalize_kernel(const float* __restrict__ partials, int nblocks,
                                float* __restrict__ out_sc) {
    __shared__ float red_h[2 * E_];
    __shared__ float red_a[2 * E_];
    __shared__ float gsum[4];
    const int t = threadIdx.x;   // 640 threads
    float asum = 0.f, hsum = 0.f;
    for (int b = 0; b < nblocks; ++b) {
        asum += partials[(size_t)b * 1280 + t];
        hsum += partials[(size_t)b * 1280 + 640 + t];
    }
    const float ap = asum / (float)M_TOK;
    const float hp = hsum / (float)M_TOK;
    red_a[t] = ap * logf(ap + 1e-7f);
    red_h[t] = hp * logf(hp + 1e-7f);
    __syncthreads();
    if (t < 2) {
        float sa = 0.f, sh = 0.f;
        for (int e = 0; e < E_; ++e) { sa += red_a[t * E_ + e]; sh += red_h[t * E_ + e]; }
        gsum[t]     = expf(-sh);   // code (hard) per-g
        gsum[2 + t] = expf(-sa);   // prob (soft) per-g
    }
    __syncthreads();
    if (t == 0) {
        out_sc[0] = gsum[0] + gsum[1];   // code_perplexity
        out_sc[1] = gsum[2] + gsum[3];   // prob_perplexity
    }
}

// ---------------------------------------------------------------------------
extern "C" void kernel_launch(void* const* d_in, const int* in_sizes, int n_in,
                              void* d_out, int out_size, void* d_ws, size_t ws_size,
                              hipStream_t stream) {
    const float* x       = (const float*)d_in[0];   // (8,2048,512)
    const float* proj_w  = (const float*)d_in[1];   // (640,512)
    const float* proj_b  = (const float*)d_in[2];   // (640,)
    const float* entries = (const float*)d_in[3];   // (1,640,384)
    const float* gumbel  = (const float*)d_in[4];   // (32768,320)

    char* ws = (char*)d_ws;
    __bf16* xhi = (__bf16*)ws;  ws += (size_t)M_TOK * K_F * sizeof(__bf16);
    __bf16* xlo = (__bf16*)ws;  ws += (size_t)M_TOK * K_F * sizeof(__bf16);
    __bf16* whi = (__bf16*)ws;  ws += (size_t)N_GE  * K_F * sizeof(__bf16);
    __bf16* wlo = (__bf16*)ws;  ws += (size_t)N_GE  * K_F * sizeof(__bf16);
    float* logits   = (float*)ws;  ws += (size_t)M_TOK * N_GE * sizeof(float);
    float* partials = (float*)ws;  // 1024 blocks * 1280 floats = 5.24 MB

    float* out_q  = (float*)d_out;                      // (16384,768)
    float* out_cb = out_q + (size_t)M_TOK * GD_;        // (16384,640)
    float* out_sc = out_cb + (size_t)M_TOK * N_GE;      // 2 scalars

    // 1) bf16 hi/lo splits
    {
        int n = M_TOK * K_F;
        split_bf16_kernel<<<(n + 255) / 256, 256, 0, stream>>>(x, xhi, xlo, n);
        int m = N_GE * K_F;
        split_bf16_kernel<<<(m + 255) / 256, 256, 0, stream>>>(proj_w, whi, wlo, m);
    }
    // 2) WMMA GEMM: 1024 row tiles x 10 col groups; 8 waves/block share cols
    {
        int nblocks = ((M_TOK / 16) / 8) * ((N_GE / 16) / 4);   // 128*10 = 1280
        gemm_logits_wmma<<<nblocks, 256, 0, stream>>>(xhi, xlo, whi, wlo, proj_b, logits);
    }
    // 3) row-wise pass: 32768 rows, 32 rows (waves) per block
    {
        int nblocks = (M_TOK * 2) / 32;                 // 1024
        row_kernel<<<nblocks, 1024, 0, stream>>>(logits, gumbel, entries,
                                                 out_q, out_cb, partials);
    }
    // 4) perplexities
    finalize_kernel<<<1, 2 * E_, 0, stream>>>(partials, (M_TOK * 2) / 32, out_sc);
}
